// RoIProcessingNetwork_77421080478032
// MI455X (gfx1250) — matrix operand order repaired and checked by reading, not compiled, not run on hardware
//
#include <hip/hip_runtime.h>
#include <hip/hip_bf16.h>
#include <math.h>

// ---------------------------------------------------------------------------
// RoI head for MI455X (gfx1250): RoIAlign -> bf16 WMMA GEMMs -> heads ->
// softmax -> decode -> per-image NMS.
// FLOP budget is dominated by FC0 (1024 x 12544 x 1024), so that (and FC1)
// run on v_wmma_f32_16x16x32_bf16 with f32 accumulation. Everything else is
// bandwidth-trivial VALU work.
// ---------------------------------------------------------------------------

typedef __bf16 bf16_t;
typedef bf16_t bf16x16 __attribute__((ext_vector_type(16)));
typedef bf16_t bf16x8  __attribute__((ext_vector_type(8)));
typedef float  f32x8   __attribute__((ext_vector_type(8)));

#define CCH   256          // channels
#define POOLP 7
#define FEAT  12544        // 256*7*7
#define HID   1024
#define NCLS  81
#define NDET  80           // NCLS-1
#define LOG_MAX_F 4.135166556742356f   // log(1000/16)

static __device__ __forceinline__ int iclamp(int v, int lo, int hi) {
  return v < lo ? lo : (v > hi ? hi : v);
}

// ---------------------------------------------------------------------------
// RoIAlign: one block per (proposal, pooled cell); thread = channel.
// Writes pooled features directly as bf16 row-major [R, 12544] (GEMM A).
// ---------------------------------------------------------------------------
__global__ __launch_bounds__(256) void roialign_kernel(
    const float* __restrict__ proposals, const int* __restrict__ imidx,
    const float* __restrict__ f0, const float* __restrict__ f1,
    const float* __restrict__ f2, const float* __restrict__ f3,
    bf16_t* __restrict__ A0, int R)
{
  int cellIdx = blockIdx.x;
  int r    = cellIdx / 49;
  int cell = cellIdx % 49;
  if (r >= R) return;
  int py = cell / POOLP, px = cell % POOLP;
  int c  = threadIdx.x;

  float x1 = proposals[r*4+0], y1 = proposals[r*4+1];
  float x2 = proposals[r*4+2], y2 = proposals[r*4+3];
  float pw = x2 - x1, ph = y2 - y1;
  float lf = floorf(4.0f + log2f(sqrtf(pw * ph) / 224.0f + 1e-6f));
  int lvl = iclamp((int)fminf(fmaxf(lf, 2.0f), 5.0f), 2, 5) - 2;

  const float* fm; int H; float inv_s;
  switch (lvl) {
    case 0:  fm = f0; H = 200; inv_s = 1.0f / 4.0f;  break;
    case 1:  fm = f1; H = 100; inv_s = 1.0f / 8.0f;  break;
    case 2:  fm = f2; H = 50;  inv_s = 1.0f / 16.0f; break;
    default: fm = f3; H = 25;  inv_s = 1.0f / 32.0f; break;
  }
  int W = H;

  float gx = ((float)px + 0.5f) / (float)POOLP;
  float gy = ((float)py + 0.5f) / (float)POOLP;
  float xs = x1 * inv_s + gx * (x2 - x1) * inv_s - 0.5f;
  float ys = y1 * inv_s + gy * (y2 - y1) * inv_s - 0.5f;
  float x0f = floorf(xs), y0f = floorf(ys);
  float wx = xs - x0f, wy = ys - y0f;
  int x0  = iclamp((int)x0f,     0, W - 1);
  int x1i = iclamp((int)x0f + 1, 0, W - 1);
  int y0  = iclamp((int)y0f,     0, H - 1);
  int y1i = iclamp((int)y0f + 1, 0, H - 1);
  int ii = imidx[r];

  size_t base = (size_t)ii * H * W;
  float v00 = fm[(base + (size_t)y0  * W + x0 ) * CCH + c];
  float v01 = fm[(base + (size_t)y0  * W + x1i) * CCH + c];
  float v10 = fm[(base + (size_t)y1i * W + x0 ) * CCH + c];
  float v11 = fm[(base + (size_t)y1i * W + x1i) * CCH + c];
  float v = v00 * (1.f - wx) * (1.f - wy) + v01 * wx * (1.f - wy)
          + v10 * (1.f - wx) * wy         + v11 * wx * wy;

  A0[(size_t)r * FEAT + (size_t)cell * CCH + c] = (bf16_t)v;
}

// ---------------------------------------------------------------------------
// Weight transpose + f32->bf16: Win[K][N] -> Wt[N][K] so GEMM B-fragment
// loads are contiguous 32B per lane.
// ---------------------------------------------------------------------------
__global__ __launch_bounds__(256) void transpose_bf16_kernel(
    const float* __restrict__ Win, bf16_t* __restrict__ Wt, int K, int N)
{
  size_t idx = (size_t)blockIdx.x * blockDim.x + threadIdx.x;
  size_t total = (size_t)K * N;
  if (idx >= total) return;
  int k = (int)(idx / N);
  int n = (int)(idx % N);
  Wt[(size_t)n * K + k] = (bf16_t)Win[idx];
}

// ---------------------------------------------------------------------------
// bf16 WMMA GEMM: C[M,N] = relu?(A[M,K] * Bt[N,K]^T + bias)
// wave32; each wave: 16(M) x 64(N) strip via 4 accumulators (A reuse).
// 8 waves/block = 64(M) x 128(N) block tile. K stepped by 32.
// A fragment (16-bit 16x32, contiguous k order): lane l, elem i ->
//   m = l%16, k = i + (l>=16 ? 16 : 0)
// B fragment (32x16): lane l, elem i -> n = l%16, k = i + (l>=16 ? 16 : 0)
// C/D: lane l, elem j -> n = l%16, m = j + (l>=16 ? 8 : 0)
// ---------------------------------------------------------------------------
__global__ __launch_bounds__(256) void gemm_bf16_wmma_kernel(
    const bf16_t* __restrict__ A,   // [M,K] row-major
    const bf16_t* __restrict__ Bt,  // [N,K] row-major (transposed weights)
    const float*  __restrict__ bias,// [N]
    float*  __restrict__ Cf32,      // optional [M,N]
    bf16_t* __restrict__ Cbf16,     // optional [M,N]
    int M, int N, int K, int relu)
{
  const int lane  = threadIdx.x & 31;
  const int wave  = threadIdx.x >> 5;
  const int waveM = wave & 3;   // 0..3 over M
  const int waveN = wave >> 2;  // 0..1 over N
  const int mBase = blockIdx.x * 64  + waveM * 16;
  const int nBase = blockIdx.y * 128 + waveN * 64;
  const int row   = lane & 15;
  const int hi    = lane >> 4;      // lane-half select

  f32x8 acc[4];
#pragma unroll
  for (int t = 0; t < 4; ++t)
#pragma unroll
    for (int j = 0; j < 8; ++j) acc[t][j] = 0.0f;

  const bf16_t* Arow = A + (size_t)(mBase + row) * K;

  for (int kk = 0; kk < K; kk += 32) {
    // A fragment: two contiguous 16B chunks per lane.
    int ka = kk + hi * 16;
    bf16x8 alo = *(const bf16x8*)(Arow + ka);
    bf16x8 ahi = *(const bf16x8*)(Arow + ka + 8);
    bf16x16 afrag;
#pragma unroll
    for (int i = 0; i < 8; ++i) { afrag[i] = alo[i]; afrag[i + 8] = ahi[i]; }

    int kb = kk + hi * 16;
#pragma unroll
    for (int t = 0; t < 4; ++t) {
      // B fragment: one contiguous 32B chunk per lane.
      const bf16_t* Brow = Bt + (size_t)(nBase + t * 16 + row) * K + kb;
      bf16x16 bfrag = *(const bf16x16*)Brow;
      acc[t] = __builtin_amdgcn_wmma_f32_16x16x32_bf16(
          /*neg_a=*/false, afrag, /*neg_b=*/false, bfrag,
          /*c_mod=*/(short)0, acc[t], /*reuse_a=*/false, /*reuse_b=*/false);
    }
  }

#pragma unroll
  for (int t = 0; t < 4; ++t) {
    int n = nBase + t * 16 + row;
    float bv = bias ? bias[n] : 0.0f;
#pragma unroll
    for (int j = 0; j < 8; ++j) {
      int m = mBase + j + hi * 8;
      float v = acc[t][j] + bv;
      if (relu) v = fmaxf(v, 0.0f);
      if (Cf32)  Cf32 [(size_t)m * N + n] = v;
      if (Cbf16) Cbf16[(size_t)m * N + n] = (bf16_t)v;
    }
  }
}

// ---------------------------------------------------------------------------
// Heads: logits[R,81] = x@Wc+bc ; reg[R,324] = x@Wr+br. One block per row,
// x row staged in LDS, threads stride the 405 output columns (coalesced
// weight reads across threads).
// ---------------------------------------------------------------------------
__global__ __launch_bounds__(256) void heads_kernel(
    const float* __restrict__ X, const float* __restrict__ Wc,
    const float* __restrict__ bc, const float* __restrict__ Wr,
    const float* __restrict__ br, float* __restrict__ logits,
    float* __restrict__ reg)
{
  __shared__ float xr[HID];
  int r = blockIdx.x;
  for (int k = threadIdx.x; k < HID; k += 256) xr[k] = X[(size_t)r * HID + k];
  __syncthreads();
  for (int c = threadIdx.x; c < NCLS + NCLS * 4; c += 256) {
    if (c < NCLS) {
      float acc = bc[c];
      for (int k = 0; k < HID; ++k) acc = fmaf(xr[k], Wc[(size_t)k * NCLS + c], acc);
      logits[(size_t)r * NCLS + c] = acc;
    } else {
      int c2 = c - NCLS;
      float acc = br[c2];
      for (int k = 0; k < HID; ++k) acc = fmaf(xr[k], Wr[(size_t)k * (NCLS * 4) + c2], acc);
      reg[(size_t)r * (NCLS * 4) + c2] = acc;
    }
  }
}

// ---------------------------------------------------------------------------
// Row softmax over 81 classes (stable).
// ---------------------------------------------------------------------------
__global__ void softmax_kernel(const float* __restrict__ logits,
                               float* __restrict__ probs, int R)
{
  int r = blockIdx.x * blockDim.x + threadIdx.x;
  if (r >= R) return;
  const float* l = logits + (size_t)r * NCLS;
  float mx = l[0];
  for (int c = 1; c < NCLS; ++c) mx = fmaxf(mx, l[c]);
  float sum = 0.0f;
  for (int c = 0; c < NCLS; ++c) sum += expf(l[c] - mx);
  float inv = 1.0f / sum;
  float* p = probs + (size_t)r * NCLS;
  for (int c = 0; c < NCLS; ++c) p[c] = expf(l[c] - mx) * inv;
}

// ---------------------------------------------------------------------------
// Box decode + validity + class-offset boxes + per-image score init.
// ---------------------------------------------------------------------------
__global__ void decode_kernel(
    const float* __restrict__ proposals, const int* __restrict__ imidx,
    const float* __restrict__ probs, const float* __restrict__ reg,
    const float* __restrict__ imsizes,
    const float* __restrict__ score_thr_p, const float* __restrict__ min_size_p,
    float* __restrict__ boxes_f, float* __restrict__ bo,
    float* __restrict__ areas, float* __restrict__ s0all,
    int R, int n)
{
  int m = blockIdx.x * blockDim.x + threadIdx.x;
  int M = R * NDET;
  if (m >= M) return;
  int r = m / NDET;
  int cls = (m % NDET) + 1;

  float x1 = proposals[r*4+0], y1 = proposals[r*4+1];
  float x2 = proposals[r*4+2], y2 = proposals[r*4+3];
  float pw = x2 - x1, ph = y2 - y1;
  float cx = (x1 + x2) * 0.5f, cy = (y1 + y2) * 0.5f;

  const float* rg = reg + (size_t)r * (NCLS * 4) + (size_t)cls * 4;
  float dx = rg[0] * 0.1f;
  float dy = rg[1] * 0.1f;
  float dw = fminf(rg[2] * 0.2f, LOG_MAX_F);
  float dh = fminf(rg[3] * 0.2f, LOG_MAX_F);
  float ncx = dx * pw + cx, ncy = dy * ph + cy;
  float nw = expf(dw) * pw, nh = expf(dh) * ph;

  int ii = imidx[r];
  float hb = imsizes[ii*2+0], wb = imsizes[ii*2+1];
  float bx1 = fminf(fmaxf(ncx - nw * 0.5f, 0.0f), wb);
  float bx2 = fminf(fmaxf(ncx + nw * 0.5f, 0.0f), wb);
  float by1 = fminf(fmaxf(ncy - nh * 0.5f, 0.0f), hb);
  float by2 = fminf(fmaxf(ncy + nh * 0.5f, 0.0f), hb);

  float scr = probs[(size_t)r * NCLS + cls];
  bool valid = (scr > *score_thr_p) && (bx2 - bx1 >= *min_size_p) &&
               (by2 - by1 >= *min_size_p);

  float mx = 0.0f;
  for (int i = 0; i < n * 2; ++i) mx = fmaxf(mx, imsizes[i]);
  float off = (float)cls * (mx + 2.0f);

  boxes_f[(size_t)m*4+0] = bx1; boxes_f[(size_t)m*4+1] = by1;
  boxes_f[(size_t)m*4+2] = bx2; boxes_f[(size_t)m*4+3] = by2;
  float o0 = bx1 + off, o1 = by1 + off, o2 = bx2 + off, o3 = by2 + off;
  bo[(size_t)m*4+0] = o0; bo[(size_t)m*4+1] = o1;
  bo[(size_t)m*4+2] = o2; bo[(size_t)m*4+3] = o3;
  areas[m] = (o2 - o0) * (o3 - o1);
  for (int i = 0; i < n; ++i)
    s0all[(size_t)i * M + m] = (valid && ii == i) ? scr : -1.0f;
}

// ---------------------------------------------------------------------------
// Greedy NMS: one 1024-thread workgroup per image. LDS argmax reduction with
// first-index tie-break (matches jnp.argmax), then parallel suppression.
// ---------------------------------------------------------------------------
__global__ __launch_bounds__(1024) void nms_kernel(
    const float* __restrict__ bo, const float* __restrict__ areas,
    const float* __restrict__ boxes_f, float* __restrict__ s0all,
    const float* __restrict__ iou_thr_p, const int* __restrict__ imtop_p,
    float* __restrict__ out, int M, int n)
{
  int img = blockIdx.x;
  float* s0 = s0all + (size_t)img * M;
  float iou_thr = *iou_thr_p;
  int imtop = *imtop_p;

  __shared__ float sv[1024];
  __shared__ int   si[1024];
  __shared__ float bb[4];
  __shared__ float bsa;
  int tid = threadIdx.x;
  int nth = blockDim.x;

  float* out_boxes = out;
  float* out_scr   = out + (size_t)n * imtop * 4;
  float* out_cls   = out + (size_t)n * imtop * 5;

  for (int t = 0; t < imtop; ++t) {
    // parallel argmax with lowest-index tie-break
    float best = -1e30f; int bidx = 0x7fffffff;
    for (int j = tid; j < M; j += nth) {
      float v = s0[j];
      if (v > best || (v == best && j < bidx)) { best = v; bidx = j; }
    }
    sv[tid] = best; si[tid] = bidx;
    __syncthreads();
    for (int s = nth >> 1; s > 0; s >>= 1) {
      if (tid < s) {
        float v2 = sv[tid + s]; int i2 = si[tid + s];
        if (v2 > sv[tid] || (v2 == sv[tid] && i2 < si[tid])) {
          sv[tid] = v2; si[tid] = i2;
        }
      }
      __syncthreads();
    }
    int i = si[0]; float s = sv[0];
    if (tid == 0) {
      bb[0] = bo[(size_t)i*4+0]; bb[1] = bo[(size_t)i*4+1];
      bb[2] = bo[(size_t)i*4+2]; bb[3] = bo[(size_t)i*4+3];
      bsa = areas[i];
      bool keep = s > 0.0f;
      size_t o = (size_t)img * imtop + t;
      out_boxes[o*4+0] = keep ? boxes_f[(size_t)i*4+0] : 0.0f;
      out_boxes[o*4+1] = keep ? boxes_f[(size_t)i*4+1] : 0.0f;
      out_boxes[o*4+2] = keep ? boxes_f[(size_t)i*4+2] : 0.0f;
      out_boxes[o*4+3] = keep ? boxes_f[(size_t)i*4+3] : 0.0f;
      out_scr[o] = keep ? s : 0.0f;
      out_cls[o] = keep ? (float)((i % NDET) + 1) : -1.0f;
    }
    __syncthreads();
    // suppress
    for (int j = tid; j < M; j += nth) {
      float iw = fmaxf(fminf(bb[2], bo[(size_t)j*4+2]) - fmaxf(bb[0], bo[(size_t)j*4+0]), 0.0f);
      float ih = fmaxf(fminf(bb[3], bo[(size_t)j*4+3]) - fmaxf(bb[1], bo[(size_t)j*4+1]), 0.0f);
      float inter = iw * ih;
      float iou = inter / (bsa + areas[j] - inter + 1e-6f);
      if (iou > iou_thr) s0[j] = -1.0f;
    }
    if (tid == 0) s0[i] = -1.0f;
    __syncthreads();
  }
}

// ---------------------------------------------------------------------------
// Host-side orchestration.
// ---------------------------------------------------------------------------
static size_t align_up(size_t x, size_t a) { return (x + a - 1) & ~(a - 1); }

extern "C" void kernel_launch(void* const* d_in, const int* in_sizes, int n_in,
                              void* d_out, int out_size, void* d_ws, size_t ws_size,
                              hipStream_t stream) {
  (void)n_in; (void)out_size;
  const float* proposals = (const float*)d_in[0];
  const int*   imidx     = (const int*)  d_in[1];
  const float* fmap0     = (const float*)d_in[2];
  const float* fmap1     = (const float*)d_in[3];
  const float* fmap2     = (const float*)d_in[4];
  const float* fmap3     = (const float*)d_in[5];
  const float* W0        = (const float*)d_in[6];
  const float* b0        = (const float*)d_in[7];
  const float* W1        = (const float*)d_in[8];
  const float* b1        = (const float*)d_in[9];
  const float* Wc        = (const float*)d_in[10];
  const float* bc        = (const float*)d_in[11];
  const float* Wr        = (const float*)d_in[12];
  const float* br        = (const float*)d_in[13];
  const float* imsizes   = (const float*)d_in[14];
  const float* score_thr = (const float*)d_in[15];
  const float* iou_thr   = (const float*)d_in[16];
  const int*   imtop     = (const int*)  d_in[17];
  const float* min_size  = (const float*)d_in[18];

  const int R = in_sizes[0] / 4;      // 1024
  const int n = in_sizes[14] / 2;     // 2
  const int M = R * NDET;             // 81920

  // Workspace carve-out (~66 MB total).
  char* base = (char*)d_ws;
  size_t off = 0;
  auto carve = [&](size_t bytes) {
    size_t o = off; off = align_up(off + bytes, 256); return (void*)(base + o);
  };
  bf16_t* A0     = (bf16_t*)carve((size_t)R * FEAT * sizeof(bf16_t));
  bf16_t* Wt0    = (bf16_t*)carve((size_t)FEAT * HID * sizeof(bf16_t));
  bf16_t* Wt1    = (bf16_t*)carve((size_t)HID * HID * sizeof(bf16_t));
  bf16_t* X1bf   = (bf16_t*)carve((size_t)R * HID * sizeof(bf16_t));
  float*  X2     = (float*) carve((size_t)R * HID * sizeof(float));
  float*  logits = (float*) carve((size_t)R * NCLS * sizeof(float));
  float*  probs  = (float*) carve((size_t)R * NCLS * sizeof(float));
  float*  regb   = (float*) carve((size_t)R * NCLS * 4 * sizeof(float));
  float*  boxesf = (float*) carve((size_t)M * 4 * sizeof(float));
  float*  bob    = (float*) carve((size_t)M * 4 * sizeof(float));
  float*  areas  = (float*) carve((size_t)M * sizeof(float));
  float*  s0all  = (float*) carve((size_t)n * M * sizeof(float));
  (void)ws_size;

  // 1) RoIAlign (writes bf16 GEMM A directly)
  roialign_kernel<<<R * 49, 256, 0, stream>>>(proposals, imidx, fmap0, fmap1,
                                              fmap2, fmap3, A0, R);

  // 2) Weight transposes (f32 [K,N] -> bf16 [N,K])
  {
    size_t tot0 = (size_t)FEAT * HID;
    transpose_bf16_kernel<<<(unsigned)((tot0 + 255) / 256), 256, 0, stream>>>(
        W0, Wt0, FEAT, HID);
    size_t tot1 = (size_t)HID * HID;
    transpose_bf16_kernel<<<(unsigned)((tot1 + 255) / 256), 256, 0, stream>>>(
        W1, Wt1, HID, HID);
  }

  // 3) FC0: [R,12544] x [12544,1024] + b0, ReLU -> bf16 X1
  {
    dim3 grid(R / 64, HID / 128);
    gemm_bf16_wmma_kernel<<<grid, 256, 0, stream>>>(
        A0, Wt0, b0, /*Cf32=*/nullptr, /*Cbf16=*/X1bf, R, HID, FEAT, /*relu=*/1);
  }

  // 4) FC1: [R,1024] x [1024,1024] + b1, ReLU -> f32 X2
  {
    dim3 grid(R / 64, HID / 128);
    gemm_bf16_wmma_kernel<<<grid, 256, 0, stream>>>(
        X1bf, Wt1, b1, /*Cf32=*/X2, /*Cbf16=*/nullptr, R, HID, HID, /*relu=*/1);
  }

  // 5) Class + box heads
  heads_kernel<<<R, 256, 0, stream>>>(X2, Wc, bc, Wr, br, logits, regb);

  // 6) Softmax
  softmax_kernel<<<(R + 127) / 128, 128, 0, stream>>>(logits, probs, R);

  // 7) Decode boxes + score init
  decode_kernel<<<(M + 255) / 256, 256, 0, stream>>>(
      proposals, imidx, probs, regb, imsizes, score_thr, min_size,
      boxesf, bob, areas, s0all, R, n);

  // 8) Per-image NMS -> outputs (boxes | scores | classes, flat)
  nms_kernel<<<n, 1024, 0, stream>>>(bob, areas, boxesf, s0all, iou_thr, imtop,
                                     (float*)d_out, M, n);
}
